// GNNAutoEncoder_10350871183824
// MI455X (gfx1250) — matrix operand order repaired and checked by reading, not compiled, not run on hardware
//
#include <hip/hip_runtime.h>
#include <hip/hip_bf16.h>
#include <stdint.h>
#include <stddef.h>

#define BLK 256
#define NSLOPE 0.01f

typedef __attribute__((ext_vector_type(16))) __bf16 v16bf;
typedef __attribute__((ext_vector_type(8)))  __bf16 v8bf;
typedef __attribute__((ext_vector_type(8)))  float  v8f;

__device__ __forceinline__ unsigned short f2bf(float f) {
  unsigned u = __float_as_uint(f);
  u += 0x7FFFu + ((u >> 16) & 1u);          // round-to-nearest-even
  return (unsigned short)(u >> 16);
}
__device__ __forceinline__ float bf2f(unsigned short b) {
  return __uint_as_float(((unsigned)b) << 16);
}
__device__ __forceinline__ float lrelu(float v) { return v > 0.f ? v : NSLOPE * v; }

// ============================ normalization ============================
__global__ void k_init_deg(float* deg, int N) {
  int i = blockIdx.x * BLK + threadIdx.x;
  if (i < N) deg[i] = 1.0f;                 // self-loop weight
}
__global__ void k_deg_edges(const int* __restrict__ dst, const float* __restrict__ w,
                            float* deg, int E) {
  int e = blockIdx.x * BLK + threadIdx.x;
  if (e < E) atomicAdd(&deg[dst[e]], w[e]);
}
__global__ void k_dinv(float* deg, float* selfn, int N) {
  int i = blockIdx.x * BLK + threadIdx.x;
  if (i >= N) return;
  float d  = deg[i];
  float di = d > 0.f ? rsqrtf(d) : 0.f;
  deg[i]   = di;                            // deg buffer becomes dinv
  selfn[i] = di * di;                       // self-loop edge norm
}
__global__ void k_norm(const int* __restrict__ src, const int* __restrict__ dst,
                       const float* __restrict__ w, const float* __restrict__ dinv,
                       float* __restrict__ nrm, int E) {
  int e = blockIdx.x * BLK + threadIdx.x;
  if (e < E) nrm[e] = dinv[src[e]] * w[e] * dinv[dst[e]];
}

// ============================ GEMMs ============================
// in: N x 3 (f32), W: 3 x 512 row-major, h: N x 512
__global__ void k_gemm_in3(const float* __restrict__ in, const float* __restrict__ W,
                           float* __restrict__ h, int N) {
  int idx = blockIdx.x * BLK + threadIdx.x;
  if (idx >= N * 512) return;
  int i = idx >> 9, f = idx & 511;
  const float* xi = in + (size_t)i * 3;
  h[idx] = xi[0] * W[f] + xi[1] * W[512 + f] + xi[2] * W[1024 + f];
}

// act: N x 512 (bf16), W: 512 x 3 row-major, h3: N x 3
__global__ void k_gemm_out3(const unsigned short* __restrict__ act,
                            const float* __restrict__ W, float* __restrict__ h3, int N) {
  int idx = blockIdx.x * BLK + threadIdx.x;
  if (idx >= N * 3) return;
  int i = idx / 3, c = idx - i * 3;
  const unsigned short* a = act + (size_t)i * 512;
  float s = 0.f;
  #pragma unroll 8
  for (int k = 0; k < 512; ++k) s += bf2f(a[k]) * W[k * 3 + c];
  h3[idx] = s;
}

// W: 512 x 512 row-major f32 -> WT: 512(col) x 512(K) bf16 (transposed)
__global__ void k_wT_bf16(const float* __restrict__ W, unsigned short* __restrict__ WT) {
  int idx = blockIdx.x * BLK + threadIdx.x;
  if (idx >= 512 * 512) return;
  int c = idx >> 9, k = idx & 511;
  WT[idx] = f2bf(W[(size_t)k * 512 + c]);   // WT[c*512+k] = W[k][c]
}

// A: N x 512 bf16 row-major activations; BT: 512(col) x 512(K) bf16; C: N x 512 f32
// One wave -> one 16(M) x 32(N) tile via two v_wmma_f32_16x16x32_bf16 per K-step.
__global__ void __launch_bounds__(BLK)
k_gemm_wmma(const unsigned short* __restrict__ A, const unsigned short* __restrict__ BT,
            float* __restrict__ C, int Nrows) {
  const int wave = (blockIdx.x * BLK + threadIdx.x) >> 5;
  const int lane = threadIdx.x & 31;
  const int NP = 16;                        // 512 / 32 column-pairs
  int mt = wave / NP, np = wave - mt * NP;
  int tilesM = (Nrows + 15) >> 4;
  if (mt >= tilesM) return;
  int m0 = mt << 4, n0 = np << 5;
  int half = lane >> 4, r = lane & 15;

  const int K = 512;
  int arowi = m0 + r; if (arowi >= Nrows) arowi = Nrows - 1;   // clamp (stores guarded)
  const unsigned short* arow = A  + (size_t)arowi * K;
  const unsigned short* b0p  = BT + (size_t)(n0 + r) * K + half * 16;
  const unsigned short* b1p  = BT + (size_t)(n0 + 16 + r) * K + half * 16;

  v8f c0 = {}; v8f c1 = {};
  for (int k0 = 0; k0 < K; k0 += 32) {
    if (k0 + 32 < K) __builtin_prefetch(arow + k0 + 32 + half * 8, 0, 1);
    // A 16x32 bf16: lanes 0-15 hold K 0-15; lanes 16-31 hold K 16-31
    v16bf a = *(const v16bf*)(arow + k0 + half * 16);
    // B 32x16 bf16: lane holds 16 contiguous K of one column (k0 + half*16 ..)
    v16bf b0 = *(const v16bf*)(b0p + k0);
    v16bf b1 = *(const v16bf*)(b1p + k0);
    c0 = __builtin_amdgcn_wmma_f32_16x16x32_bf16(false, a, false, b0, (short)0, c0, false, false);
    c1 = __builtin_amdgcn_wmma_f32_16x16x32_bf16(false, a, false, b1, (short)0, c1, false, false);
  }
  // C/D layout: VGPR v -> row m0 + half*8 + v, col n0 + (lane&15)
  int rbase = m0 + half * 8;
  #pragma unroll
  for (int v = 0; v < 8; ++v) {
    int row = rbase + v;
    if (row < Nrows) {
      float* crow = C + (size_t)row * 512 + n0 + r;
      crow[0]  = c0[v];
      crow[16] = c1[v];
    }
  }
}

// ============================ aggregation ============================
__global__ void k_agg_init512(const float* __restrict__ h, const float* __restrict__ selfn,
                              float* __restrict__ agg, int N) {
  int idx = blockIdx.x * BLK + threadIdx.x;
  if (idx >= N * 512) return;
  agg[idx] = h[idx] * selfn[idx >> 9];
}
__global__ void k_agg_init3(const float* __restrict__ h3, const float* __restrict__ selfn,
                            float* __restrict__ agg3, int N) {
  int idx = blockIdx.x * BLK + threadIdx.x;
  if (idx >= N * 3) return;
  agg3[idx] = h3[idx] * selfn[idx / 3];
}
// one wave covers 128 contiguous features of one edge -> coalesced 512B gather
__global__ void k_agg_edges512(const int* __restrict__ src, const int* __restrict__ dst,
                               const float* __restrict__ nrm, const float* __restrict__ h,
                               float* __restrict__ agg, int E) {
  int idx = blockIdx.x * BLK + threadIdx.x;
  if (idx >= E * 128) return;
  int e = idx >> 7, fo = (idx & 127) << 2;
  int s = src[e], d = dst[e];
  float w = nrm[e];
  const float4 v = *(const float4*)(h + (size_t)s * 512 + fo);
  float* o = agg + (size_t)d * 512 + fo;
  atomicAdd(o + 0, v.x * w);
  atomicAdd(o + 1, v.y * w);
  atomicAdd(o + 2, v.z * w);
  atomicAdd(o + 3, v.w * w);
}
__global__ void k_agg_edges3(const int* __restrict__ src, const int* __restrict__ dst,
                             const float* __restrict__ nrm, const float* __restrict__ h3,
                             float* __restrict__ agg3, int E) {
  int e = blockIdx.x * BLK + threadIdx.x;
  if (e >= E) return;
  int s = src[e], d = dst[e];
  float w = nrm[e];
  const float* hv = h3 + (size_t)s * 3;
  float* o = agg3 + (size_t)d * 3;
  atomicAdd(o + 0, hv[0] * w);
  atomicAdd(o + 1, hv[1] * w);
  atomicAdd(o + 2, hv[2] * w);
}
__global__ void k_fin512(const float* __restrict__ agg, const float* __restrict__ b,
                         unsigned short* __restrict__ actb, int N) {
  int idx = blockIdx.x * BLK + threadIdx.x;
  if (idx >= N * 512) return;
  actb[idx] = f2bf(lrelu(agg[idx] + b[idx & 511]));
}
__global__ void k_fin3_relu(const float* __restrict__ agg3, const float* __restrict__ b,
                            float* __restrict__ act3, int N) {
  int idx = blockIdx.x * BLK + threadIdx.x;
  if (idx >= N * 3) return;
  act3[idx] = lrelu(agg3[idx] + b[idx % 3]);
}
__global__ void k_fin3_out(const float* __restrict__ agg3, const float* __restrict__ b,
                           float* __restrict__ out, int N) {
  int idx = blockIdx.x * BLK + threadIdx.x;
  if (idx >= N * 3) return;
  out[idx] = agg3[idx] + b[idx % 3];        // final layer: no lrelu
}

// ============================ launcher ============================
static inline unsigned gsz(long long n) { return (unsigned)((n + BLK - 1) / BLK); }

extern "C" void kernel_launch(void* const* d_in, const int* in_sizes, int n_in,
                              void* d_out, int out_size, void* d_ws, size_t ws_size,
                              hipStream_t stream) {
  const float* x  = (const float*)d_in[0];
  const int*   ei = (const int*)d_in[1];
  const float* ew = (const float*)d_in[2];
  const float* W1 = (const float*)d_in[3];  const float* b1 = (const float*)d_in[4];
  const float* W2 = (const float*)d_in[5];  const float* b2 = (const float*)d_in[6];
  const float* W3 = (const float*)d_in[7];  const float* b3 = (const float*)d_in[8];
  const float* W4 = (const float*)d_in[9];  const float* b4 = (const float*)d_in[10];
  const float* W5 = (const float*)d_in[11]; const float* b5 = (const float*)d_in[12];
  const float* W6 = (const float*)d_in[13]; const float* b6 = (const float*)d_in[14];

  const int N = in_sizes[0] / 3;
  const int E = in_sizes[2];
  const int H = 512;
  const int* srcv = ei;
  const int* dstv = ei + E;

  // workspace carve (256B aligned)
  char* p = (char*)d_ws;
  auto carve = [&](size_t bytes) -> char* {
    char* r = p; p += (bytes + 255) & ~(size_t)255; return r;
  };
  float*          dinv  = (float*)carve((size_t)N * 4);        // deg -> dinv
  float*          selfn = (float*)carve((size_t)N * 4);
  float*          nrm   = (float*)carve((size_t)E * 4);
  float*          h     = (float*)carve((size_t)N * H * 4);    // GEMM output
  float*          agg   = (float*)carve((size_t)N * H * 4);    // scatter accumulator
  unsigned short* actb  = (unsigned short*)carve((size_t)N * H * 2); // bf16 activations
  unsigned short* WbT   = (unsigned short*)carve((size_t)H * H * 2); // transposed bf16 W
  float*          h3    = (float*)carve((size_t)N * 3 * 4);
  float*          agg3  = (float*)carve((size_t)N * 3 * 4);
  float*          act3  = (float*)carve((size_t)N * 3 * 4);
  (void)ws_size; (void)n_in; (void)out_size;

  const unsigned gN   = gsz(N);
  const unsigned gE   = gsz(E);
  const unsigned gNH  = gsz((long long)N * H);
  const unsigned gN3  = gsz((long long)N * 3);
  const unsigned gEF  = gsz((long long)E * 128);
  const unsigned gW   = gsz((long long)H * H);
  const int tilesM = (N + 15) / 16;
  const unsigned gMM = gsz((long long)tilesM * 16 * 32);       // waves * 32 threads

  // --- gcn_norm (once; reused by all layers) ---
  k_init_deg <<<gN, BLK, 0, stream>>>(dinv, N);
  k_deg_edges<<<gE, BLK, 0, stream>>>(dstv, ew, dinv, E);
  k_dinv     <<<gN, BLK, 0, stream>>>(dinv, selfn, N);
  k_norm     <<<gE, BLK, 0, stream>>>(srcv, dstv, ew, dinv, nrm, E);

  // --- layer 1: 3 -> 512 ---
  k_gemm_in3    <<<gNH, BLK, 0, stream>>>(x, W1, h, N);
  k_agg_init512 <<<gNH, BLK, 0, stream>>>(h, selfn, agg, N);
  k_agg_edges512<<<gEF, BLK, 0, stream>>>(srcv, dstv, nrm, h, agg, E);
  k_fin512      <<<gNH, BLK, 0, stream>>>(agg, b1, actb, N);

  // --- layer 2: 512 -> 512 (WMMA) ---
  k_wT_bf16     <<<gW,  BLK, 0, stream>>>(W2, WbT);
  k_gemm_wmma   <<<gMM, BLK, 0, stream>>>(actb, WbT, h, N);
  k_agg_init512 <<<gNH, BLK, 0, stream>>>(h, selfn, agg, N);
  k_agg_edges512<<<gEF, BLK, 0, stream>>>(srcv, dstv, nrm, h, agg, E);
  k_fin512      <<<gNH, BLK, 0, stream>>>(agg, b2, actb, N);

  // --- layer 3: 512 -> 3 ---
  k_gemm_out3 <<<gN3, BLK, 0, stream>>>(actb, W3, h3, N);
  k_agg_init3 <<<gN3, BLK, 0, stream>>>(h3, selfn, agg3, N);
  k_agg_edges3<<<gE,  BLK, 0, stream>>>(srcv, dstv, nrm, h3, agg3, E);
  k_fin3_relu <<<gN3, BLK, 0, stream>>>(agg3, b3, act3, N);

  // --- layer 4: 3 -> 512 ---
  k_gemm_in3    <<<gNH, BLK, 0, stream>>>(act3, W4, h, N);
  k_agg_init512 <<<gNH, BLK, 0, stream>>>(h, selfn, agg, N);
  k_agg_edges512<<<gEF, BLK, 0, stream>>>(srcv, dstv, nrm, h, agg, E);
  k_fin512      <<<gNH, BLK, 0, stream>>>(agg, b4, actb, N);

  // --- layer 5: 512 -> 512 (WMMA) ---
  k_wT_bf16     <<<gW,  BLK, 0, stream>>>(W5, WbT);
  k_gemm_wmma   <<<gMM, BLK, 0, stream>>>(actb, WbT, h, N);
  k_agg_init512 <<<gNH, BLK, 0, stream>>>(h, selfn, agg, N);
  k_agg_edges512<<<gEF, BLK, 0, stream>>>(srcv, dstv, nrm, h, agg, E);
  k_fin512      <<<gNH, BLK, 0, stream>>>(agg, b5, actb, N);

  // --- layer 6: 512 -> 3 (no activation) ---
  k_gemm_out3 <<<gN3, BLK, 0, stream>>>(actb, W6, h3, N);
  k_agg_init3 <<<gN3, BLK, 0, stream>>>(h3, selfn, agg3, N);
  k_agg_edges3<<<gE,  BLK, 0, stream>>>(srcv, dstv, nrm, h3, agg3, E);
  k_fin3_out  <<<gN3, BLK, 0, stream>>>(agg3, b6, (float*)d_out, N);
}
